// GatedDeltaNet_25331717111964
// MI455X (gfx1250) — matrix operand order repaired and checked by reading, not compiled, not run on hardware
//
#include <hip/hip_runtime.h>
#include <hip/hip_bf16.h>

// ---------------------------------------------------------------------------
// Gated DeltaNet forward for MI455X (gfx1250, wave32)
//   B=4, T=2048, D=1024, H=16, DK=64, DV=128, QKD=1024, VD=2048, KW=4
// GEMMs: v_wmma_f32_16x16x32_bf16 with double-buffered ASYNC LDS staging
// (global_load_async_to_lds_b128 + s_wait_asynccnt). bf16 operands are
// pre-converted / pre-transposed once. The delta-rule scan keeps the 64x128
// fp32 state in registers per (b,h) block.
// ---------------------------------------------------------------------------

#define Bc 4
#define Tc 2048
#define Dc 1024
#define Hc 16
#define DKc 64
#define DVc 128
#define Mrows (Bc * Tc)       // 8192
#define EPSc 1e-5f

typedef __attribute__((ext_vector_type(16))) __bf16 v16bf;
typedef __attribute__((ext_vector_type(8)))  float  v8f;

typedef union { unsigned int u[8]; v16bf v; } FragBF;

__device__ __forceinline__ float sigmoidf_(float x) {
    return 1.0f / (1.0f + __expf(-x));
}

// Async DMA: 16B per lane, global -> LDS, tracked by ASYNCcnt.
// VDST carries the LDS byte offset (low 32 bits of the generic shared addr).
__device__ __forceinline__ void async_copy_b128(const void* gptr, void* lptr) {
    const unsigned long long ga = (unsigned long long)gptr;
    const unsigned int       la = (unsigned int)(unsigned long long)lptr;
    asm volatile("global_load_async_to_lds_b128 %0, %1, off"
                 :: "v"(la), "v"(ga)
                 : "memory");
}

__device__ __forceinline__ void wait_async0() {
#if __has_builtin(__builtin_amdgcn_s_wait_asynccnt)
    __builtin_amdgcn_s_wait_asynccnt(0);
#else
    asm volatile("s_wait_asynccnt 0x0" ::: "memory");
#endif
}

// ---------------------------------------------------------------------------
// One-shot fp32 -> bf16 conversion (grid-stride).
// ---------------------------------------------------------------------------
__global__ __launch_bounds__(256)
void gdn_cvt_bf16(const float* __restrict__ in, __bf16* __restrict__ out, long n) {
    for (long i = (long)blockIdx.x * 256 + threadIdx.x; i < n; i += (long)gridDim.x * 256)
        out[i] = (__bf16)in[i];
}

// fp32 [K][N] -> bf16 transposed [N][K]
__global__ __launch_bounds__(256)
void gdn_cvt_tr_bf16(const float* __restrict__ src, __bf16* __restrict__ dst,
                     int K, int N) {
    const long total = (long)K * N;
    for (long i = (long)blockIdx.x * 256 + threadIdx.x; i < total; i += (long)gridDim.x * 256) {
        const int n = (int)(i / K);
        const int k = (int)(i % K);
        dst[i] = (__bf16)src[(long)k * N + n];
    }
}

// ---------------------------------------------------------------------------
// WMMA GEMM: C[M,N] = A[M,K] * Bt[N,K]^T   (A, Bt bf16; C fp32)
// optional epilogue C *= sigmoid(gate).
// Block tile 64(M) x 128(N), 256 threads = 8 waves; wave = 16(M) x 64(N)
// strip = 4 accumulators sharing one A fragment. K staged 32/iter via async
// DMA into double-buffered LDS. Requires M%64==0, N%128==0, K%32==0.
// ---------------------------------------------------------------------------
#define LDSP 40   // ushort row stride: mult of 8 -> 16B-aligned b128 transfers

__global__ __launch_bounds__(256)
void gdn_wmma_gemm(const __bf16* __restrict__ A, const __bf16* __restrict__ Bt,
                   float* __restrict__ C, int M, int N, int K,
                   const float* __restrict__ gate) {
    __shared__ unsigned short As[2][64 * LDSP];
    __shared__ unsigned short Bs[2][128 * LDSP];

    const int tid  = threadIdx.x;
    const int lane = tid & 31;
    const int wav  = tid >> 5;          // 0..7
    const int tm   = wav & 3;           // 4 m-tiles of 16
    const int tng  = wav >> 2;          // 0..1 : n-group of 4 tiles
    const int m0   = blockIdx.y * 64;
    const int n0   = blockIdx.x * 128;

    const int halfw = lane >> 4;        // 0 | 1
    const int l15   = lane & 15;

    // cooperative-copy coords: each thread moves 3 x b128 (8 bf16) per stage
    const int ld_row = tid >> 2;        // 0..63
    const int ld_k8  = (tid & 3) * 8;   // 0,8,16,24

    auto issue_stage = [&](int k0, int buf) {
        async_copy_b128(&A [(size_t)(m0 + ld_row)      * K + k0 + ld_k8],
                        &As[buf][ld_row * LDSP + ld_k8]);
        async_copy_b128(&Bt[(size_t)(n0 + ld_row)      * K + k0 + ld_k8],
                        &Bs[buf][ld_row * LDSP + ld_k8]);
        async_copy_b128(&Bt[(size_t)(n0 + 64 + ld_row) * K + k0 + ld_k8],
                        &Bs[buf][(64 + ld_row) * LDSP + ld_k8]);
    };

    v8f acc[4] = {};
    issue_stage(0, 0);

    int buf = 0;
    for (int k0 = 0; k0 < K; k0 += 32) {
        wait_async0();       // own wave's stage-s DMA landed
        __syncthreads();     // all waves' stage-s visible; prev reads of buf^1 done
        if (k0 + 32 < K) issue_stage(k0 + 32, buf ^ 1);   // overlap with WMMAs

        // A fragment per ISA 16-bit A(16x32) lane layout
        FragBF fa;
        const int mrow = tm * 16 + l15;
        #pragma unroll
        for (int j = 0; j < 8; ++j) {
            const int kb = (j < 4) ? (halfw * 8 + 2 * j)
                                   : (16 + halfw * 8 + 2 * (j - 4));
            fa.u[j] = *(const unsigned int*)&As[buf][mrow * LDSP + kb];
        }
        // 4 B fragments / 4 WMMAs reuse fa
        #pragma unroll
        for (int t = 0; t < 4; ++t) {
            FragBF fb;
            const int ncol = (tng * 4 + t) * 16 + l15;
            #pragma unroll
            for (int j = 0; j < 8; ++j) {
                fb.u[j] = *(const unsigned int*)&Bs[buf][ncol * LDSP + ((j < 4) ? (halfw * 8 + 2 * j) : (16 + halfw * 8 + 2 * (j - 4)))];
            }
            acc[t] = __builtin_amdgcn_wmma_f32_16x16x32_bf16(
                         false, fa.v, false, fb.v, (short)0, acc[t], false, false);
        }
        buf ^= 1;
    }

    // epilogue: C/D layout -> VGPR r holds (m = half*8 + r, n = l15)
    #pragma unroll
    for (int t = 0; t < 4; ++t) {
        const int n = n0 + (tng * 4 + t) * 16 + l15;
        #pragma unroll
        for (int r = 0; r < 8; ++r) {
            const int m = m0 + tm * 16 + halfw * 8 + r;
            float val = acc[t][r];
            if (gate != nullptr) {
                val *= sigmoidf_(gate[(size_t)m * N + n]);
            }
            C[(size_t)m * N + n] = val;
        }
    }
}

// ---------------------------------------------------------------------------
// Zero-centered RMS norm over last axis (in place). One block per row.
// ---------------------------------------------------------------------------
__global__ __launch_bounds__(256)
void gdn_zcrms(float* __restrict__ x, const float* __restrict__ scale, int Cn) {
    float* xr = x + (size_t)blockIdx.x * Cn;
    __shared__ float r1[256], r2[256];
    float s1 = 0.f, s2 = 0.f;
    for (int c = threadIdx.x; c < Cn; c += 256) {
        const float v = xr[c];
        s1 += v; s2 += v * v;
    }
    r1[threadIdx.x] = s1; r2[threadIdx.x] = s2;
    __syncthreads();
    for (int off = 128; off > 0; off >>= 1) {
        if (threadIdx.x < off) {
            r1[threadIdx.x] += r1[threadIdx.x + off];
            r2[threadIdx.x] += r2[threadIdx.x + off];
        }
        __syncthreads();
    }
    const float mean = r1[0] / (float)Cn;
    const float var  = r2[0] / (float)Cn - mean * mean;
    const float inv  = rsqrtf(var + EPSc);
    for (int c = threadIdx.x; c < Cn; c += 256) {
        xr[c] = (xr[c] - mean) * inv * scale[c];
    }
}

// ---------------------------------------------------------------------------
// Causal depthwise conv (K=4) + SiLU.
// ---------------------------------------------------------------------------
__global__ __launch_bounds__(256)
void gdn_conv_silu(const float* __restrict__ in, const float* __restrict__ kern,
                   const float* __restrict__ bias, float* __restrict__ out,
                   int Cn, long total) {
    const long idx = (long)blockIdx.x * 256 + threadIdx.x;
    if (idx >= total) return;
    const int  c = (int)(idx % Cn);
    const long r = idx / Cn;             // global row b*T + t
    const int  t = (int)(r % Tc);
    float acc = bias[c];
    #pragma unroll
    for (int i = 0; i < 4; ++i) {
        const int ts = t - 3 + i;
        if (ts >= 0) acc += in[(r - 3 + i) * Cn + c] * kern[i * Cn + c];
    }
    out[idx] = acc * sigmoidf_(acc);     // SiLU
}

// ---------------------------------------------------------------------------
// alpha/beta: sigmoid(x @ W) with W [D,16]. One block per row; 16 lanes/head.
// ---------------------------------------------------------------------------
__global__ __launch_bounds__(256)
void gdn_alphabeta(const float* __restrict__ x,
                   const float* __restrict__ aw, const float* __restrict__ bw,
                   float* __restrict__ a, float* __restrict__ b) {
    const int r = blockIdx.x;
    const int h = threadIdx.x >> 4;
    const int lane = threadIdx.x & 15;
    const float* xr = x + (size_t)r * Dc;
    float pa = 0.f, pb = 0.f;
    for (int k = lane; k < Dc; k += 16) {
        const float xv = xr[k];
        pa += xv * aw[k * Hc + h];
        pb += xv * bw[k * Hc + h];
    }
    __shared__ float ra[256], rb[256];
    ra[threadIdx.x] = pa; rb[threadIdx.x] = pb;
    __syncthreads();
    for (int off = 8; off > 0; off >>= 1) {
        if (lane < off) {
            ra[threadIdx.x] += ra[threadIdx.x + off];
            rb[threadIdx.x] += rb[threadIdx.x + off];
        }
        __syncthreads();
    }
    if (lane == 0) {
        a[(size_t)r * Hc + h] = sigmoidf_(ra[threadIdx.x]);
        b[(size_t)r * Hc + h] = sigmoidf_(rb[threadIdx.x]);
    }
}

// ---------------------------------------------------------------------------
// Delta-rule scan. 64 blocks (one per (b,h)), 256 threads.
// Thread (col = tid>>1, half = tid&1) owns S[half*32 .. +32)[col] in regs.
// Pair partial sums combined with wave32 __shfl_xor(.,1).
// Output written directly as bf16 (only consumer is the final WMMA GEMM).
// ---------------------------------------------------------------------------
#define SCH 8   // time steps staged per LDS chunk

__global__ __launch_bounds__(256)
void gdn_scan(const float* __restrict__ q, const float* __restrict__ k,
              const float* __restrict__ v, const float* __restrict__ alpha,
              const float* __restrict__ beta, __bf16* __restrict__ o) {
    const int bh = blockIdx.x;
    const int b  = bh / Hc;
    const int h  = bh % Hc;
    const int tid  = threadIdx.x;
    const int col  = tid >> 1;      // 0..127  (DV)
    const int half = tid & 1;       // DK half
    const int dk0  = half * 32;

    __shared__ float shq[SCH][DKc], shk[SCH][DKc], shv[SCH][DVc];
    __shared__ float sha[SCH], shb[SCH];

    float s[32];
    #pragma unroll
    for (int i = 0; i < 32; ++i) s[i] = 0.f;

    for (int tc = 0; tc < Tc; tc += SCH) {
        __syncthreads();
        for (int i = tid; i < SCH * DKc; i += 256) {
            const int tt = i >> 6, dd = i & 63;
            const size_t base = ((size_t)(b * Tc + tc + tt)) * (Hc * DKc) + h * DKc + dd;
            shq[tt][dd] = q[base];
            shk[tt][dd] = k[base];
        }
        for (int i = tid; i < SCH * DVc; i += 256) {
            const int tt = i >> 7, dd = i & 127;
            shv[tt][dd] = v[((size_t)(b * Tc + tc + tt)) * (Hc * DVc) + h * DVc + dd];
        }
        if (tid < SCH) {
            sha[tid] = alpha[(size_t)(b * Tc + tc + tid) * Hc + h];
        } else if (tid < 2 * SCH) {
            const int tt = tid - SCH;
            shb[tt] = beta[(size_t)(b * Tc + tc + tt) * Hc + h];
        }
        __syncthreads();

        #pragma unroll
        for (int ts = 0; ts < SCH; ++ts) {
            const float at = sha[ts];
            const float bt = shb[ts];
            // pred = a * (k . S[:,col])  (pair-split over DK)
            float p = 0.f;
            #pragma unroll
            for (int i = 0; i < 32; ++i) p += shk[ts][dk0 + i] * s[i];
            p += __shfl_xor(p, 1, 32);
            const float delta = bt * (shv[ts][col] - at * p);
            // S = a*S + k (x) delta ; o = q . S
            float oacc = 0.f;
            #pragma unroll
            for (int i = 0; i < 32; ++i) {
                s[i] = at * s[i] + shk[ts][dk0 + i] * delta;
                oacc += shq[ts][dk0 + i] * s[i];
            }
            oacc += __shfl_xor(oacc, 1, 32);
            if (half == 0) {
                o[((size_t)(b * Tc + tc + ts)) * (Hc * DVc) + h * DVc + col] = (__bf16)oacc;
            }
        }
    }
}

// ---------------------------------------------------------------------------
// Launch
// ---------------------------------------------------------------------------
extern "C" void kernel_launch(void* const* d_in, const int* in_sizes, int n_in,
                              void* d_out, int out_size, void* d_ws, size_t ws_size,
                              hipStream_t stream) {
    const float* x        = (const float*)d_in[0];
    const float* q_w      = (const float*)d_in[1];
    const float* k_w      = (const float*)d_in[2];
    const float* v_w      = (const float*)d_in[3];
    const float* q_scale  = (const float*)d_in[4];
    const float* k_scale  = (const float*)d_in[5];
    const float* v_scale  = (const float*)d_in[6];
    const float* q_conv_k = (const float*)d_in[7];
    const float* q_conv_b = (const float*)d_in[8];
    const float* k_conv_k = (const float*)d_in[9];
    const float* k_conv_b = (const float*)d_in[10];
    const float* v_conv_k = (const float*)d_in[11];
    const float* v_conv_b = (const float*)d_in[12];
    const float* alpha_w  = (const float*)d_in[13];
    const float* beta_w   = (const float*)d_in[14];
    const float* out_w    = (const float*)d_in[15];
    const float* gate_w   = (const float*)d_in[16];
    float* out            = (float*)d_out;

    // ---- fp32 workspace ----
    float* ws = (float*)d_ws;
    const size_t sz1 = (size_t)Mrows * 1024;   // 8 Mi floats
    const size_t sz2 = (size_t)Mrows * 2048;
    const size_t szh = (size_t)Mrows * Hc;

    float* q_pre = ws;                       // later reused as gate_pre
    float* k_pre = ws + sz1;
    float* v_pre = ws + 2 * sz1;
    float* qb    = ws + 2 * sz1 + sz2;
    float* kb    = qb + sz1;
    float* vb    = kb + sz1;
    float* ab    = vb + sz2;
    float* bb    = ab + szh;
    float* f32_end = bb + szh;
    float* gate_pre = q_pre;

    // ---- bf16 workspace (after fp32 region) ----
    __bf16* bw0 = (__bf16*)f32_end;
    __bf16* xb    = bw0;                                   // 8 Mi  [8192][1024]
    __bf16* wqt   = xb  + sz1;                             // 1 Mi  [1024][1024]
    __bf16* wkt   = wqt + (size_t)1024 * 1024;             // 1 Mi
    __bf16* wgt   = wkt + (size_t)1024 * 1024;             // 1 Mi
    __bf16* wvt   = wgt + (size_t)1024 * 1024;             // 2 Mi  [2048][1024]
    __bf16* wot   = wvt + (size_t)2048 * 1024;             // 2 Mi  [1024][2048]
    __bf16* coreb = wot + (size_t)1024 * 2048;             // 16 Mi [8192][2048]

    const dim3 blk(256);
    const long totQ = (long)Mrows * 1024;
    const long totV = (long)Mrows * 2048;

    // ---- one-shot bf16 conversions / weight transposes ----
    gdn_cvt_bf16<<<4096, blk, 0, stream>>>(x, xb, totQ);
    gdn_cvt_tr_bf16<<<1024, blk, 0, stream>>>(q_w,    wqt, 1024, 1024);
    gdn_cvt_tr_bf16<<<1024, blk, 0, stream>>>(k_w,    wkt, 1024, 1024);
    gdn_cvt_tr_bf16<<<1024, blk, 0, stream>>>(gate_w, wgt, 1024, 1024);
    gdn_cvt_tr_bf16<<<2048, blk, 0, stream>>>(v_w,    wvt, 1024, 2048);  // [1024][2048] -> [2048][1024]
    gdn_cvt_tr_bf16<<<2048, blk, 0, stream>>>(out_w,  wot, 2048, 1024);  // [2048][1024] -> [1024][2048]

    // ---- q branch ----
    gdn_wmma_gemm<<<dim3(1024 / 128, Mrows / 64), blk, 0, stream>>>(
        xb, wqt, q_pre, Mrows, 1024, 1024, nullptr);
    gdn_zcrms<<<Mrows, blk, 0, stream>>>(q_pre, q_scale, 1024);
    gdn_conv_silu<<<(totQ + 255) / 256, blk, 0, stream>>>(
        q_pre, q_conv_k, q_conv_b, qb, 1024, totQ);

    // ---- k branch ----
    gdn_wmma_gemm<<<dim3(1024 / 128, Mrows / 64), blk, 0, stream>>>(
        xb, wkt, k_pre, Mrows, 1024, 1024, nullptr);
    gdn_zcrms<<<Mrows, blk, 0, stream>>>(k_pre, k_scale, 1024);
    gdn_conv_silu<<<(totQ + 255) / 256, blk, 0, stream>>>(
        k_pre, k_conv_k, k_conv_b, kb, 1024, totQ);

    // ---- v branch ----
    gdn_wmma_gemm<<<dim3(2048 / 128, Mrows / 64), blk, 0, stream>>>(
        xb, wvt, v_pre, Mrows, 2048, 1024, nullptr);
    gdn_zcrms<<<Mrows, blk, 0, stream>>>(v_pre, v_scale, 2048);
    gdn_conv_silu<<<(totV + 255) / 256, blk, 0, stream>>>(
        v_pre, v_conv_k, v_conv_b, vb, 2048, totV);

    // ---- gates ----
    gdn_alphabeta<<<Mrows, blk, 0, stream>>>(x, alpha_w, beta_w, ab, bb);

    // ---- recurrent delta-rule scan -> bf16 core ----
    gdn_scan<<<Bc * Hc, blk, 0, stream>>>(qb, kb, vb, ab, bb, coreb);

    // ---- output: (core @ out_w) * sigmoid(x @ gate_w) ----
    gdn_wmma_gemm<<<dim3(1024 / 128, Mrows / 64), blk, 0, stream>>>(
        xb, wgt, gate_pre, Mrows, 1024, 1024, nullptr);
    gdn_wmma_gemm<<<dim3(1024 / 128, Mrows / 64), blk, 0, stream>>>(
        coreb, wot, out, Mrows, 1024, 2048, gate_pre);
}
